// GATNet_35150012351302
// MI455X (gfx1250) — compile-verified, matrix operand order checked by reading, not run on hardware
//
#include <hip/hip_runtime.h>

typedef __attribute__((ext_vector_type(16))) _Float16 v16h;
typedef __attribute__((ext_vector_type(8)))  _Float16 v8h;
typedef __attribute__((ext_vector_type(8)))  float    v8f;

#define N1        150000
#define N1T       50000
#define E1        2000000
#define N2T       10000
#define E2        400000
#define F_IN      512
#define D1        64
#define NHEAD     8
#define CHEAD     8
#define CLS       41
#define NEG_SLOPE 0.2f

__device__ __forceinline__ float lrelu(float x) { return x > 0.f ? x : NEG_SLOPE * x; }

// float atomic max via sign-split int/uint atomics (monotone bit ordering)
__device__ __forceinline__ void atomicMaxF(float* addr, float v) {
  if (v >= 0.f) atomicMax((int*)addr, __float_as_int(v));
  else          atomicMin((unsigned int*)addr, __float_as_uint(v));
}

// ---------------- utility: fill ----------------
__global__ void k_fill(float* __restrict__ p, long n, float v) {
  long i = (long)blockIdx.x * blockDim.x + threadIdx.x;
  if (i < n) p[i] = v;
}

// ---------------- W1^T -> f16 [64][512] ----------------
__global__ void k_w1t(const float* __restrict__ W1, _Float16* __restrict__ Wt) {
  int i = blockIdx.x * blockDim.x + threadIdx.x;
  if (i >= D1 * F_IN) return;
  int n = i / F_IN, k = i - n * F_IN;
  Wt[i] = (_Float16)W1[(size_t)k * D1 + n];
}

// ---------------- GEMM1: H1[150000,64] = f16(x[b1_n_id]) @ f16(W1), wave-per-16-rows, WMMA ----------------
__global__ void k_gemm1(const float* __restrict__ x, const int* __restrict__ nid,
                        const _Float16* __restrict__ Wt, float* __restrict__ H1) {
  const int ntiles = N1 / 16;
  int wave = blockIdx.x * (blockDim.x >> 5) + (threadIdx.x >> 5);
  if (wave >= ntiles) return;               // wave-uniform guard: EXEC stays all-ones for WMMA
  int lane = threadIdx.x & 31;
  int lo = lane & 15, hi = lane >> 4;
  const float* ap = x + (size_t)nid[wave * 16 + lo] * F_IN + hi * 8;  // fused gather

  v8f acc[4] = {};
  for (int k0 = 0; k0 < F_IN; k0 += 32) {
    const float* a0 = ap + k0;
    float4 f0 = *(const float4*)(a0 + 0);
    float4 f1 = *(const float4*)(a0 + 4);
    float4 f2 = *(const float4*)(a0 + 16);
    float4 f3 = *(const float4*)(a0 + 20);
    v16h a;
    a[0]=(_Float16)f0.x;  a[1]=(_Float16)f0.y;  a[2]=(_Float16)f0.z;  a[3]=(_Float16)f0.w;
    a[4]=(_Float16)f1.x;  a[5]=(_Float16)f1.y;  a[6]=(_Float16)f1.z;  a[7]=(_Float16)f1.w;
    a[8]=(_Float16)f2.x;  a[9]=(_Float16)f2.y;  a[10]=(_Float16)f2.z; a[11]=(_Float16)f2.w;
    a[12]=(_Float16)f3.x; a[13]=(_Float16)f3.y; a[14]=(_Float16)f3.z; a[15]=(_Float16)f3.w;
#pragma unroll
    for (int nf = 0; nf < 4; nf++) {
      const _Float16* bp = Wt + (size_t)(nf * 16 + lo) * F_IN + k0 + hi * 8;
      v8h b0 = *(const v8h*)(bp);
      v8h b1 = *(const v8h*)(bp + 16);
      v16h b;
#pragma unroll
      for (int i = 0; i < 8; i++) { b[i] = b0[i]; b[i + 8] = b1[i]; }
      acc[nf] = __builtin_amdgcn_wmma_f32_16x16x32_f16(
          false, a, false, b, (short)0, acc[nf], false, false);
    }
  }
  float* op = H1 + (size_t)wave * 16 * D1;
#pragma unroll
  for (int nf = 0; nf < 4; nf++)
#pragma unroll
    for (int r = 0; r < 8; r++)
      op[(size_t)(hi * 8 + r) * D1 + nf * 16 + lo] = acc[nf][r];
}

// ---------------- per-node attention logits, 8 heads ----------------
__global__ void k_att1(const float* __restrict__ H1, const float* __restrict__ avec,
                       const int* __restrict__ gather, float* __restrict__ out, int n) {
  int i = blockIdx.x * blockDim.x + threadIdx.x;
  if (i >= n) return;
  int node = gather ? gather[i] : i;
  const float* h = H1 + (size_t)node * D1;
#pragma unroll
  for (int hh = 0; hh < NHEAD; hh++) {
    float s = 0.f;
#pragma unroll
    for (int c = 0; c < CHEAD; c++) s += h[hh * CHEAD + c] * avec[hh * CHEAD + c];
    out[(size_t)i * NHEAD + hh] = s;
  }
}

// ---------------- block-0 edge passes ----------------
__global__ void k_e1_max(const int* __restrict__ src, const int* __restrict__ dst,
                         const float* __restrict__ as, const float* __restrict__ ad,
                         float* __restrict__ m, int ne) {
  int e = blockIdx.x * blockDim.x + threadIdx.x;
  if (e >= ne) return;
  int s = src[e], d = dst[e];
#pragma unroll
  for (int h = 0; h < NHEAD; h++) {
    float v = lrelu(as[(size_t)s * NHEAD + h] + ad[(size_t)d * NHEAD + h]);
    atomicMaxF(&m[(size_t)d * NHEAD + h], v);
  }
}

__global__ void k_e1_sum(const int* __restrict__ src, const int* __restrict__ dst,
                         const float* __restrict__ as, const float* __restrict__ ad,
                         const float* __restrict__ m, float* __restrict__ ssum, int ne) {
  int e = blockIdx.x * blockDim.x + threadIdx.x;
  if (e >= ne) return;
  int s = src[e], d = dst[e];
#pragma unroll
  for (int h = 0; h < NHEAD; h++) {
    float v = lrelu(as[(size_t)s * NHEAD + h] + ad[(size_t)d * NHEAD + h]);
    atomicAdd(&ssum[(size_t)d * NHEAD + h], __expf(v - m[(size_t)d * NHEAD + h]));
  }
}

__global__ void k_e1_agg(const int* __restrict__ src, const int* __restrict__ dst,
                         const float* __restrict__ as, const float* __restrict__ ad,
                         const float* __restrict__ m, const float* __restrict__ ssum,
                         const float* __restrict__ H1, float* __restrict__ out, int ne) {
  int e = blockIdx.x * blockDim.x + threadIdx.x;
  if (e >= ne) return;
  int s = src[e], d = dst[e];
  float al[NHEAD];
#pragma unroll
  for (int h = 0; h < NHEAD; h++) {
    float v = lrelu(as[(size_t)s * NHEAD + h] + ad[(size_t)d * NHEAD + h]);
    al[h] = __expf(v - m[(size_t)d * NHEAD + h]) / (ssum[(size_t)d * NHEAD + h] + 1e-16f);
  }
  const float* hs = H1 + (size_t)s * D1;
  float* od = out + (size_t)d * D1;
#pragma unroll
  for (int c = 0; c < D1; c++) atomicAdd(&od[c], hs[c] * al[c >> 3]);
}

// ---------------- bias + ELU (in place) ----------------
__global__ void k_bias_elu(float* __restrict__ o, const float* __restrict__ bias, long n) {
  long i = (long)blockIdx.x * blockDim.x + threadIdx.x;
  if (i >= n) return;
  float v = o[i] + bias[i & (D1 - 1)];
  o[i] = v > 0.f ? v : (__expf(v) - 1.f);
}

// ---------------- GEMM2: H2[50000,41] = h1 @ W2 (tiny, scalar FMA) ----------------
__global__ void k_gemm2(const float* __restrict__ h1, const float* __restrict__ W2,
                        float* __restrict__ H2) {
  int i = blockIdx.x * blockDim.x + threadIdx.x;
  if (i >= N1T * CLS) return;
  int n = i / CLS, j = i - n * CLS;
  const float* hr = h1 + (size_t)n * D1;
  float s = 0.f;
#pragma unroll 8
  for (int k = 0; k < D1; k++) s += hr[k] * W2[k * CLS + j];
  H2[i] = s;
}

// ---------------- per-node attention logits, 1 head of 41 ----------------
__global__ void k_att2(const float* __restrict__ H2, const float* __restrict__ avec,
                       const int* __restrict__ gather, float* __restrict__ out, int n) {
  int i = blockIdx.x * blockDim.x + threadIdx.x;
  if (i >= n) return;
  int node = gather ? gather[i] : i;
  const float* h = H2 + (size_t)node * CLS;
  float s = 0.f;
#pragma unroll
  for (int c = 0; c < CLS; c++) s += h[c] * avec[c];
  out[i] = s;
}

// ---------------- block-1 edge passes ----------------
__global__ void k_e2_max(const int* __restrict__ src, const int* __restrict__ dst,
                         const float* __restrict__ as, const float* __restrict__ ad,
                         float* __restrict__ m, int ne) {
  int e = blockIdx.x * blockDim.x + threadIdx.x;
  if (e >= ne) return;
  float v = lrelu(as[src[e]] + ad[dst[e]]);
  atomicMaxF(&m[dst[e]], v);
}

__global__ void k_e2_sum(const int* __restrict__ src, const int* __restrict__ dst,
                         const float* __restrict__ as, const float* __restrict__ ad,
                         const float* __restrict__ m, float* __restrict__ ssum, int ne) {
  int e = blockIdx.x * blockDim.x + threadIdx.x;
  if (e >= ne) return;
  int d = dst[e];
  float v = lrelu(as[src[e]] + ad[d]);
  atomicAdd(&ssum[d], __expf(v - m[d]));
}

__global__ void k_e2_agg(const int* __restrict__ src, const int* __restrict__ dst,
                         const float* __restrict__ as, const float* __restrict__ ad,
                         const float* __restrict__ m, const float* __restrict__ ssum,
                         const float* __restrict__ H2, float* __restrict__ out, int ne) {
  int e = blockIdx.x * blockDim.x + threadIdx.x;
  if (e >= ne) return;
  int s = src[e], d = dst[e];
  float v = lrelu(as[s] + ad[d]);
  float al = __expf(v - m[d]) / (ssum[d] + 1e-16f);
  const float* hs = H2 + (size_t)s * CLS;
  float* od = out + (size_t)d * CLS;
#pragma unroll
  for (int c = 0; c < CLS; c++) atomicAdd(&od[c], hs[c] * al);
}

// ---------------- bias + log_softmax ----------------
__global__ void k_lsm(const float* __restrict__ o2, const float* __restrict__ bias,
                      float* __restrict__ out) {
  int t = blockIdx.x * blockDim.x + threadIdx.x;
  if (t >= N2T) return;
  float z[CLS];
  float mx = -3.4e38f;
#pragma unroll
  for (int c = 0; c < CLS; c++) { z[c] = o2[(size_t)t * CLS + c] + bias[c]; mx = fmaxf(mx, z[c]); }
  float se = 0.f;
#pragma unroll
  for (int c = 0; c < CLS; c++) se += __expf(z[c] - mx);
  float l = __logf(se) + mx;
#pragma unroll
  for (int c = 0; c < CLS; c++) out[(size_t)t * CLS + c] = z[c] - l;
}

extern "C" void kernel_launch(void* const* d_in, const int* in_sizes, int n_in,
                              void* d_out, int out_size, void* d_ws, size_t ws_size,
                              hipStream_t stream) {
  (void)in_sizes; (void)n_in; (void)out_size; (void)ws_size;
  const float* x     = (const float*)d_in[0];
  const float* W1    = (const float*)d_in[1];
  const float* a1s   = (const float*)d_in[2];
  const float* a1d   = (const float*)d_in[3];
  const float* bias1 = (const float*)d_in[4];
  const float* W2    = (const float*)d_in[5];
  const float* a2sw  = (const float*)d_in[6];
  const float* a2dw  = (const float*)d_in[7];
  const float* bias2 = (const float*)d_in[8];
  const int* b1_nid  = (const int*)d_in[9];
  const int* b1_res  = (const int*)d_in[10];
  const int* b1_src  = (const int*)d_in[11];
  const int* b1_dst  = (const int*)d_in[12];
  const int* b2_res  = (const int*)d_in[13];
  const int* b2_src  = (const int*)d_in[14];
  const int* b2_dst  = (const int*)d_in[15];
  float* out = (float*)d_out;

  char* ws = (char*)d_ws;
  size_t off = 0;
  auto alloc = [&](size_t b) { size_t r = off; off += (b + 255) & ~(size_t)255; return r; };
  _Float16* Wt1 = (_Float16*)(ws + alloc((size_t)D1 * F_IN * 2));
  float* H1  = (float*)(ws + alloc((size_t)N1 * D1 * 4));
  float* AS1 = (float*)(ws + alloc((size_t)N1 * NHEAD * 4));
  float* AD1 = (float*)(ws + alloc((size_t)N1T * NHEAD * 4));
  float* M1  = (float*)(ws + alloc((size_t)N1T * NHEAD * 4));
  float* S1  = (float*)(ws + alloc((size_t)N1T * NHEAD * 4));
  float* O1  = (float*)(ws + alloc((size_t)N1T * D1 * 4));
  float* H2  = (float*)(ws + alloc((size_t)N1T * CLS * 4));
  float* A2S = (float*)(ws + alloc((size_t)N1T * 4));
  float* A2D = (float*)(ws + alloc((size_t)N2T * 4));
  float* M2  = (float*)(ws + alloc((size_t)N2T * 4));
  float* S2  = (float*)(ws + alloc((size_t)N2T * 4));
  float* O2  = (float*)(ws + alloc((size_t)N2T * CLS * 4));

  const int B = 256;
  auto G = [](long n, int b) { return (unsigned)((n + b - 1) / b); };
  const float NEG_INF = -__builtin_inff();

  k_w1t<<<G(D1 * F_IN, B), B, 0, stream>>>(W1, Wt1);
  {
    int waves = N1 / 16, wpb = 4;
    k_gemm1<<<(waves + wpb - 1) / wpb, wpb * 32, 0, stream>>>(x, b1_nid, Wt1, H1);
  }
  k_att1<<<G(N1, B), B, 0, stream>>>(H1, a1s, nullptr, AS1, N1);
  k_att1<<<G(N1T, B), B, 0, stream>>>(H1, a1d, b1_res, AD1, N1T);
  k_fill<<<G((long)N1T * NHEAD, B), B, 0, stream>>>(M1, (long)N1T * NHEAD, NEG_INF);
  k_fill<<<G((long)N1T * NHEAD, B), B, 0, stream>>>(S1, (long)N1T * NHEAD, 0.f);
  k_fill<<<G((long)N1T * D1, B), B, 0, stream>>>(O1, (long)N1T * D1, 0.f);
  k_e1_max<<<G(E1, B), B, 0, stream>>>(b1_src, b1_dst, AS1, AD1, M1, E1);
  k_e1_sum<<<G(E1, B), B, 0, stream>>>(b1_src, b1_dst, AS1, AD1, M1, S1, E1);
  k_e1_agg<<<G(E1, B), B, 0, stream>>>(b1_src, b1_dst, AS1, AD1, M1, S1, H1, O1, E1);
  k_bias_elu<<<G((long)N1T * D1, B), B, 0, stream>>>(O1, bias1, (long)N1T * D1);
  k_gemm2<<<G((long)N1T * CLS, B), B, 0, stream>>>(O1, W2, H2);
  k_att2<<<G(N1T, B), B, 0, stream>>>(H2, a2sw, nullptr, A2S, N1T);
  k_att2<<<G(N2T, B), B, 0, stream>>>(H2, a2dw, b2_res, A2D, N2T);
  k_fill<<<G(N2T, B), B, 0, stream>>>(M2, N2T, NEG_INF);
  k_fill<<<G(N2T, B), B, 0, stream>>>(S2, N2T, 0.f);
  k_fill<<<G((long)N2T * CLS, B), B, 0, stream>>>(O2, (long)N2T * CLS, 0.f);
  k_e2_max<<<G(E2, B), B, 0, stream>>>(b2_src, b2_dst, A2S, A2D, M2, E2);
  k_e2_sum<<<G(E2, B), B, 0, stream>>>(b2_src, b2_dst, A2S, A2D, M2, S2, E2);
  k_e2_agg<<<G(E2, B), B, 0, stream>>>(b2_src, b2_dst, A2S, A2D, M2, S2, H2, O2, E2);
  k_lsm<<<G(N2T, B), B, 0, stream>>>(O2, bias2, out);
}
